// MultiheadSelfAttention_5265629905563
// MI455X (gfx1250) — matrix operand order repaired and checked by reading, not compiled, not run on hardware
//
#include <hip/hip_runtime.h>

// ---------------- problem constants ----------------
#define T_SEQ 2048
#define BATCH 2
#define EMB   1024
#define NHEAD 16
#define HDIM  64
#define NTOK  (T_SEQ * BATCH)   // 4096 tokens, token m = t*B + b (row-major [T,B,E])
#define N3E   (3 * EMB)         // 3072

typedef __attribute__((ext_vector_type(16))) __bf16 bf16x16;
typedef __attribute__((ext_vector_type(8)))  float  f32x8;

union FragAB { bf16x16 v; float4 f4[2]; };

__device__ __forceinline__ float4 ld16(const __bf16* p) {
  return *reinterpret_cast<const float4*>(p);
}

__device__ __forceinline__ f32x8 wmma_bf16(bf16x16 a, bf16x16 b, f32x8 c) {
  // D = A(16x32 bf16) * B(32x16 bf16) + C(16x16 f32)
  return __builtin_amdgcn_wmma_f32_16x16x32_bf16(false, a, false, b, (short)0, c, false, false);
}

__device__ __forceinline__ float redmax16(float v) {
#pragma unroll
  for (int m = 1; m < 16; m <<= 1) v = fmaxf(v, __shfl_xor(v, m, 32));
  return v;
}
__device__ __forceinline__ float redsum16(float v) {
#pragma unroll
  for (int m = 1; m < 16; m <<= 1) v += __shfl_xor(v, m, 32);
  return v;
}

// A 16x32 bf16 fragment: lane = row, interleaved K halves {0..7,16..23}/{8..15,24..31}
__device__ __forceinline__ void load_a(FragAB& a, const __bf16* arow, int k0, int lh) {
  a.f4[0] = ld16(arow + k0 + lh * 8);
  a.f4[1] = ld16(arow + k0 + 16 + lh * 8);
}
// B 32x16 bf16 fragment: lane = col, interleaved K halves {0..7,16..23}/{8..15,24..31}
__device__ __forceinline__ void load_b(FragAB& bm, const __bf16* brow, int k0, int lh) {
  bm.f4[0] = ld16(brow + k0 + lh * 8);
  bm.f4[1] = ld16(brow + k0 + 16 + lh * 8);
}

// ---------------- f32 -> bf16 convert ----------------
__global__ void cvt_f32_bf16(const float* __restrict__ in, __bf16* __restrict__ out, int n) {
  int i = blockIdx.x * blockDim.x + threadIdx.x;
  if (i < n) out[i] = (__bf16)in[i];
}

// ---------------- shared GEMM mainloop: one wave -> 16(M) x 64(N) tile ----------------
// act: [M][1024] bf16 row-major, W: [N][1024] bf16 row-major, C = act * W^T
// Software-pipelined: fragments for step k+32 are loaded before the WMMAs of step k.
__device__ __forceinline__ void gemm_tile_16x64(const __bf16* __restrict__ act,
                                                const __bf16* __restrict__ W,
                                                int m0, int n0, f32x8 acc[4],
                                                int li, int lh) {
  const __bf16* arow = act + (long)(m0 + li) * EMB;
  const __bf16* brow[4];
#pragma unroll
  for (int t = 0; t < 4; ++t) brow[t] = W + (long)(n0 + t * 16 + li) * EMB;

  FragAB a[2];
  FragAB bm[2][4];
  load_a(a[0], arow, 0, lh);
#pragma unroll
  for (int t = 0; t < 4; ++t) load_b(bm[0][t], brow[t], 0, lh);

#pragma unroll 2
  for (int k0 = 0; k0 < EMB - 32; k0 += 32) {
    const int cur = (k0 >> 5) & 1;
    const int nxt = cur ^ 1;
    load_a(a[nxt], arow, k0 + 32, lh);           // prefetch next step
#pragma unroll
    for (int t = 0; t < 4; ++t) load_b(bm[nxt][t], brow[t], k0 + 32, lh);
#pragma unroll
    for (int t = 0; t < 4; ++t) acc[t] = wmma_bf16(a[cur].v, bm[cur][t].v, acc[t]);
  }
  const int last = ((EMB - 64) >> 5 & 1) ^ 1;    // buffer holding the final k-step
#pragma unroll
  for (int t = 0; t < 4; ++t) acc[t] = wmma_bf16(a[last].v, bm[last][t].v, acc[t]);
}

// ---------------- in-projection: proj = q @ W_in^T + b_in ; scatter to per-head Q,K,V^T ----------------
__global__ __launch_bounds__(256) void proj_kernel(const __bf16* __restrict__ act,
                                                   const __bf16* __restrict__ Win,
                                                   const float* __restrict__ bin,
                                                   __bf16* __restrict__ qh,   // [BH][T][64]
                                                   __bf16* __restrict__ kh,   // [BH][T][64]
                                                   __bf16* __restrict__ vT) { // [BH][64][T]
  const int lane = threadIdx.x & 31, wave = threadIdx.x >> 5;
  const int li = lane & 15, lh = lane >> 4;
  const int m0 = blockIdx.y * 128 + wave * 16;
  const int n0 = blockIdx.x * 64;
  f32x8 acc[4] = {};
  gemm_tile_16x64(act, Win, m0, n0, acc, li, lh);
#pragma unroll
  for (int t = 0; t < 4; ++t) {
    const int j = n0 + t * 16 + li;        // output column (lane-dependent)
    const float bv = bin[j];
    const int part = j >> 10;              // 0=q 1=k 2=v (uniform within a tile)
    const int e = j & 1023;
    const int h = e >> 6, hd = e & 63;
#pragma unroll
    for (int r = 0; r < 8; ++r) {          // C layout: VGPR r -> row lh*8+r
      const int m = m0 + lh * 8 + r;       // token index = t_tok*BATCH + b
      const int tt = m >> 1, b = m & 1;
      const int bh = b * NHEAD + h;
      const float val = acc[t][r] + bv;
      if (part == 0)
        qh[((long)bh * T_SEQ + tt) * HDIM + hd] = (__bf16)(val * 0.125f);
      else if (part == 1)
        kh[((long)bh * T_SEQ + tt) * HDIM + hd] = (__bf16)val;
      else
        vT[((long)bh * HDIM + hd) * T_SEQ + tt] = (__bf16)val;
    }
  }
}

// ---------------- fused flash attention ----------------
// grid: x = BH (32), y = T/128 (16). 8 waves/block, each wave = 16 query rows.
__global__ __launch_bounds__(256) void attn_kernel(const __bf16* __restrict__ qh,
                                                   const __bf16* __restrict__ kh,
                                                   const __bf16* __restrict__ vT,
                                                   const unsigned char* __restrict__ mask, // [B][T], 1 = pad
                                                   __bf16* __restrict__ ctx) {             // [NTOK][EMB]
  __shared__ __bf16 Psh[8][16][72];   // per-wave P slab, row stride 144B (16B aligned)
  const int lane = threadIdx.x & 31, wave = threadIdx.x >> 5;
  const int li = lane & 15, lh = lane >> 4;
  const int bh = blockIdx.x;
  const int b = bh >> 4, h = bh & 15;
  const int rowBase = blockIdx.y * 128 + wave * 16;

  // Q fragments (16 rows x 64 hd) kept in registers for the whole key loop
  FragAB qa[2];
  const __bf16* qrow = qh + ((long)bh * T_SEQ + rowBase + li) * HDIM;
  load_a(qa[0], qrow, 0, lh);
  load_a(qa[1], qrow, 32, lh);

  float m_i[8], l_i[8];
  f32x8 acc[4] = {};
#pragma unroll
  for (int r = 0; r < 8; ++r) { m_i[r] = -INFINITY; l_i[r] = 0.0f; }

  const unsigned char* mrow = mask + b * T_SEQ;

  for (int s0 = 0; s0 < T_SEQ; s0 += 64) {
    if (mrow[s0]) break;                 // padding is monotone: rest of keys are pad

    // ---- issue ALL K-fragment loads for this chunk, then all 8 QK^T WMMAs ----
    FragAB kb[4][2];
#pragma unroll
    for (int t = 0; t < 4; ++t) {
      const __bf16* krow = kh + ((long)bh * T_SEQ + s0 + t * 16 + li) * HDIM;
      load_b(kb[t][0], krow, 0, lh);
      load_b(kb[t][1], krow, 32, lh);
    }
    f32x8 sc[4];
#pragma unroll
    for (int t = 0; t < 4; ++t) {
      f32x8 c = {};
      c = wmma_bf16(qa[0].v, kb[t][0].v, c);
      c = wmma_bf16(qa[1].v, kb[t][1].v, c);
      if (mrow[s0 + t * 16 + li]) {      // per-lane key-padding mask (lane = key col)
#pragma unroll
        for (int r = 0; r < 8; ++r) c[r] = -INFINITY;
      }
      sc[t] = c;
    }

    // ---- online softmax update (rows live on VGPR index, cols across 16 lanes) ----
    float mn[8], al[8];
#pragma unroll
    for (int r = 0; r < 8; ++r) {
      float cm = fmaxf(fmaxf(sc[0][r], sc[1][r]), fmaxf(sc[2][r], sc[3][r]));
      cm = redmax16(cm);
      mn[r] = fmaxf(m_i[r], cm);
      al[r] = __expf(m_i[r] - mn[r]);
    }
    float rs[8] = {0, 0, 0, 0, 0, 0, 0, 0};
#pragma unroll
    for (int t = 0; t < 4; ++t) {
#pragma unroll
      for (int r = 0; r < 8; ++r) {
        float p = __expf(sc[t][r] - mn[r]);
        rs[r] += p;
        Psh[wave][lh * 8 + r][t * 16 + li] = (__bf16)p;  // C layout -> row-major P
      }
    }
#pragma unroll
    for (int r = 0; r < 8; ++r) {
      rs[r] = redsum16(rs[r]);
      l_i[r] = l_i[r] * al[r] + rs[r];
      m_i[r] = mn[r];
    }
#pragma unroll
    for (int t = 0; t < 4; ++t)
#pragma unroll
      for (int r = 0; r < 8; ++r) acc[t][r] *= al[r];

    __builtin_amdgcn_wave_barrier();     // keep DS store->load order (slab is wave-private)

    // ---- issue ALL V-fragment (global) and P-fragment (LDS) loads, then 8 PV WMMAs ----
    FragAB vb[4][2];
#pragma unroll
    for (int t = 0; t < 4; ++t) {
      const __bf16* vrow = vT + ((long)bh * HDIM + t * 16 + li) * T_SEQ + s0;
      load_b(vb[t][0], vrow, 0, lh);
      load_b(vb[t][1], vrow, 32, lh);
    }
    const __bf16* prow = &Psh[wave][li][0];
    FragAB pa[2];
    load_a(pa[0], prow, 0, lh);
    load_a(pa[1], prow, 32, lh);
#pragma unroll
    for (int t = 0; t < 4; ++t) {
      acc[t] = wmma_bf16(pa[0].v, vb[t][0].v, acc[t]);
      acc[t] = wmma_bf16(pa[1].v, vb[t][1].v, acc[t]);
    }
  }

  // ---- normalize and write context [token][E] as bf16 ----
  float inv[8];
#pragma unroll
  for (int r = 0; r < 8; ++r) inv[r] = 1.0f / l_i[r];
#pragma unroll
  for (int t = 0; t < 4; ++t)
#pragma unroll
    for (int r = 0; r < 8; ++r) {
      const int tt = rowBase + lh * 8 + r;
      const float val = acc[t][r] * inv[r];
      ctx[((long)(tt * BATCH + b)) * EMB + h * HDIM + t * 16 + li] = (__bf16)val;
    }
}

// ---------------- out-projection: out = ctx @ W_out^T + b_out (f32 output) ----------------
__global__ __launch_bounds__(256) void outproj_kernel(const __bf16* __restrict__ act,
                                                      const __bf16* __restrict__ Wout,
                                                      const float* __restrict__ bout,
                                                      float* __restrict__ out) {
  const int lane = threadIdx.x & 31, wave = threadIdx.x >> 5;
  const int li = lane & 15, lh = lane >> 4;
  const int m0 = blockIdx.y * 128 + wave * 16;
  const int n0 = blockIdx.x * 64;
  f32x8 acc[4] = {};
  gemm_tile_16x64(act, Wout, m0, n0, acc, li, lh);
#pragma unroll
  for (int t = 0; t < 4; ++t) {
    const int j = n0 + t * 16 + li;
    const float bv = bout[j];
#pragma unroll
    for (int r = 0; r < 8; ++r) {
      const int m = m0 + lh * 8 + r;
      out[(long)m * EMB + j] = acc[t][r] + bv;
    }
  }
}

// ---------------- launch ----------------
extern "C" void kernel_launch(void* const* d_in, const int* in_sizes, int n_in,
                              void* d_out, int out_size, void* d_ws, size_t ws_size,
                              hipStream_t stream) {
  const float* query = (const float*)d_in[0];
  const unsigned char* mask = (const unsigned char*)d_in[1];  // bool [B][T]
  const float* W_in  = (const float*)d_in[2];
  const float* b_in  = (const float*)d_in[3];
  const float* W_out = (const float*)d_in[4];
  const float* b_out = (const float*)d_in[5];
  float* out = (float*)d_out;
  (void)in_sizes; (void)n_in; (void)out_size; (void)ws_size;

  char* ws = (char*)d_ws;
  // workspace layout (40 MB total): ctx reuses the query-bf16 region (proj finishes first)
  __bf16* qbf   = (__bf16*)(ws);                        //  8 MB: query bf16 / later ctx
  __bf16* Winb  = (__bf16*)(ws + (8u  << 20));          //  6 MB
  __bf16* Woutb = (__bf16*)(ws + (14u << 20));          //  2 MB
  __bf16* qh    = (__bf16*)(ws + (16u << 20));          //  8 MB  [BH][T][64]
  __bf16* kh    = (__bf16*)(ws + (24u << 20));          //  8 MB  [BH][T][64]
  __bf16* vT    = (__bf16*)(ws + (32u << 20));          //  8 MB  [BH][64][T]
  __bf16* ctx   = qbf;

  const int nq = NTOK * EMB;        // 4194304
  const int nwi = N3E * EMB;        // 3145728
  const int nwo = EMB * EMB;        // 1048576
  cvt_f32_bf16<<<(nq  + 255) / 256, 256, 0, stream>>>(query, qbf, nq);
  cvt_f32_bf16<<<(nwi + 255) / 256, 256, 0, stream>>>(W_in, Winb, nwi);
  cvt_f32_bf16<<<(nwo + 255) / 256, 256, 0, stream>>>(W_out, Woutb, nwo);

  proj_kernel<<<dim3(N3E / 64, NTOK / 128), 256, 0, stream>>>(qbf, Winb, b_in, qh, kh, vT);
  attn_kernel<<<dim3(BATCH * NHEAD, T_SEQ / 128), 256, 0, stream>>>(qh, kh, vT, mask, ctx);
  outproj_kernel<<<dim3(EMB / 64, NTOK / 128), 256, 0, stream>>>(ctx, Woutb, b_out, out);
}
